// Test_3461743640652
// MI455X (gfx1250) — compile-verified, matrix operand order checked by reading, not run on hardware
//
#include <hip/hip_runtime.h>
#include <hip/hip_bf16.h>

// ---------------------------------------------------------------------------
// Types for CDNA5 WMMA (wave32, 16x16x32 bf16 -> f32)
// ---------------------------------------------------------------------------
typedef __attribute__((ext_vector_type(16))) __bf16 v16bf;
typedef __attribute__((ext_vector_type(8)))  __bf16 v8bf;
typedef __attribute__((ext_vector_type(8)))  float  v8f;

#define Bk 64
#define Lk 128
#define Tk 127
#define Dk 300
#define Sk 256

__device__ __forceinline__ __bf16 to_bf(float f) { return (__bf16)f; }

// K-pattern for 16-bit A/B fragments (ISA 7.12.2):
// lane<16: K = e<8 ? e : e+8 ; lane>=16: +8
__device__ __forceinline__ int frag_koff(int lane) { return (lane >> 4) << 3; }

// Load a 16x32 bf16 fragment from a row-major matrix (ld elements per row).
// For the A matrix: row0 = M-tile base. For B used as Bt (N,K) row-major:
// row0 = N-tile base. Requires ld % 8 == 0 and k0 % 8 == 0 (16B aligned).
__device__ __forceinline__ v16bf load_frag_bf16(const __bf16* p, int ld,
                                                int row0, int k0) {
  int lane = threadIdx.x & 31;
  const __bf16* base =
      p + (size_t)(row0 + (lane & 15)) * ld + k0 + frag_koff(lane);
  v8bf lo = *(const v8bf*)(base);
  v8bf hi = *(const v8bf*)(base + 16);
  v16bf r;
#pragma unroll
  for (int e = 0; e < 8; ++e) { r[e] = lo[e]; r[e + 8] = hi[e]; }
  return r;
}

__device__ __forceinline__ v8f wmma_bf16(v16bf a, v16bf b, v8f c) {
  return __builtin_amdgcn_wmma_f32_16x16x32_bf16(false, a, false, b,
                                                 (short)0, c, false, false);
}

__device__ __forceinline__ float sigmoidf_(float x) {
  return 1.0f / (1.0f + __expf(-x));
}

// Async copy of one time-step's input block (64 rows x 304 bf16, 38 x 16B
// chunks per row) from global bf16 to an LDS buffer via the CDNA5 async
// data path (GLOBAL_LOAD_ASYNC_TO_LDS_B128, tracked by ASYNCcnt).
__device__ __forceinline__ void async_copy_x(const __bf16* __restrict__ xg,
                                             __bf16* dstbuf, int t) {
  for (int c = threadIdx.x; c < 64 * 38; c += blockDim.x) {
    int m = c / 38, j = c % 38;
    const __bf16* g = xg + ((size_t)m * Sk + t) * 304 + j * 8;
    unsigned ldsoff = (unsigned)(uintptr_t)(dstbuf + m * 608 + j * 8);
    // vdst = LDS byte address, vaddr = 64-bit global address
    asm volatile("global_load_async_to_lds_b128 %0, %1, off"
                 :: "v"(ldsoff), "v"(g) : "memory");
  }
}

__device__ __forceinline__ void wait_async0() {
  asm volatile("s_wait_asynccnt 0x0" ::: "memory");
}

// ---------------------------------------------------------------------------
// Utility kernels
// ---------------------------------------------------------------------------
__global__ void zero_bf16_kernel(__bf16* p, size_t n) {
  for (size_t i = (size_t)blockIdx.x * blockDim.x + threadIdx.x; i < n;
       i += (size_t)gridDim.x * blockDim.x)
    p[i] = to_bf(0.0f);
}

// LSTM kernel src (600,1200) f32 -> dst bf16 transposed, per-gate padded:
// dst shape (4*304, 608). K layout matches LDS operand [x 0..299 | pad
// 300..303 | h 304..603 | pad 604..607]:
//   kp<300        -> src row kp          (x part)
//   304<=kp<604   -> src row kp-4        (h part)
//   else          -> 0
__global__ void conv_lstm_w_kernel(const float* __restrict__ src,
                                   __bf16* __restrict__ dst) {
  const size_t total = (size_t)1216 * 608;
  for (size_t idx = (size_t)blockIdx.x * blockDim.x + threadIdx.x; idx < total;
       idx += (size_t)gridDim.x * blockDim.x) {
    int row = (int)(idx / 608);
    int kp = (int)(idx % 608);
    int q = row / 304, n = row % 304;
    int k = (kp < 300) ? kp : (kp >= 304 && kp < 604) ? (kp - 4) : -1;
    float v = (n < 300 && k >= 0) ? src[(size_t)k * 1200 + q * 300 + n] : 0.f;
    dst[idx] = to_bf(v);
  }
}

// dec_w (2400,300) f32 -> dst bf16 transposed padded (304, 2400)
__global__ void conv_dec_w_kernel(const float* __restrict__ src,
                                  __bf16* __restrict__ dst) {
  const size_t total = (size_t)304 * 2400;
  for (size_t idx = (size_t)blockIdx.x * blockDim.x + threadIdx.x; idx < total;
       idx += (size_t)gridDim.x * blockDim.x) {
    int n = (int)(idx / 2400);
    int k = (int)(idx % 2400);
    float v = (n < 300) ? src[(size_t)k * 300 + n] : 0.f;
    dst[idx] = to_bf(v);
  }
}

// f32 (rows, 300) -> bf16 (rows, 304) zero-padded (LSTM input staging format)
__global__ void conv_x_kernel(const float* __restrict__ src,
                              __bf16* __restrict__ dst) {
  const size_t total = (size_t)Bk * Sk * 304;
  for (size_t idx = (size_t)blockIdx.x * blockDim.x + threadIdx.x; idx < total;
       idx += (size_t)gridDim.x * blockDim.x) {
    size_t row = idx / 304;
    int n = (int)(idx % 304);
    dst[idx] = to_bf((n < 300) ? src[row * 300 + n] : 0.f);
  }
}

// tree[b,s,:] = 0 for s==0 or s>L, else embed[x[b,s-1]]
__global__ void embed_tree_kernel(const int* __restrict__ x,
                                  const float* __restrict__ emb,
                                  float* __restrict__ tree) {
  int row = blockIdx.x;           // b*256 + s
  int b = row >> 8, s = row & 255;
  float* dst = tree + (size_t)row * Dk;
  if (s >= 1 && s <= Lk) {
    const float* e = emb + (size_t)x[b * Lk + (s - 1)] * Dk;
    for (int k = threadIdx.x; k < Dk; k += blockDim.x) dst[k] = e[k];
  } else {
    for (int k = threadIdx.x; k < Dk; k += blockDim.x) dst[k] = 0.f;
  }
}

// Recursive tree unfold: one block per batch element, sequential i=1..T-1.
__global__ void unfold_kernel(float* __restrict__ tree,
                              const int* __restrict__ temp,
                              const int* __restrict__ tag,
                              const int* __restrict__ len,
                              const float* __restrict__ tcw,
                              const float* __restrict__ tpw) {
  __shared__ float cemb[600];
  __shared__ float tvec[16];
  int b = blockIdx.x;
  int tid = threadIdx.x;
  int ln = len[b];
  for (int i = 1; i < Tk; ++i) {
    int c0 = temp[((size_t)b * Tk + i) * 2 + 0];
    int c1 = temp[((size_t)b * Tk + i) * 2 + 1];
    for (int k = tid; k < 600; k += blockDim.x)
      cemb[k] = (k < 300) ? tree[((size_t)b * Sk + c0) * Dk + k]
                          : tree[((size_t)b * Sk + c1) * Dk + (k - 300)];
    __syncthreads();
    int t0 = (c0 == 0) ? 0 : tag[(size_t)b * 255 + c0 - 1];
    int t1 = (c1 == 0) ? 0 : tag[(size_t)b * 255 + c1 - 1];
    int pp = ln + 1 + i;
    int pt = (pp == 0) ? 0 : tag[(size_t)b * 255 + pp - 1];
    if (tid < 10) {
      float s = 0.f;
      for (int k = 0; k < 600; ++k) {
        float w = (k < 300) ? tcw[(size_t)t0 * 3000 + k * 10 + tid]
                            : tcw[(size_t)t1 * 3000 + (k - 300) * 10 + tid];
        s += w * cemb[k];
      }
      tvec[tid] = s;
    }
    __syncthreads();
    for (int j = tid; j < Dk; j += blockDim.x) {
      float s = 0.f;
#pragma unroll
      for (int f = 0; f < 10; ++f)
        s += tpw[(size_t)pt * 3000 + j * 10 + f] * tvec[f];
      tree[((size_t)b * Sk + pp) * Dk + j] += s;
    }
    __threadfence_block();
    __syncthreads();
  }
}

// ---------------------------------------------------------------------------
// Bidirectional LSTM stage. grid=4: blockIdx = {seq0 fw, seq0 bw, seq1 fw,
// seq1 bw}. One 512-thread workgroup per pass; recurrent state lives in LDS.
// x: bf16 (B, S, 304) zero-padded. Weights: bf16 (1216, 608) per-gate padded,
// transposed, K layout [x|pad|h|pad]. The [x_t|h] operand is double-buffered
// in LDS; x_{t+1} is staged with async-to-LDS copies overlapped with step t's
// WMMA work, and h_t is written into the next buffer.
// ---------------------------------------------------------------------------
__global__ void __launch_bounds__(512) lstm_kernel(
    const __bf16* __restrict__ xA, const __bf16* __restrict__ xB,
    const __bf16* __restrict__ wtF, const float* __restrict__ bF,
    const __bf16* __restrict__ wtB, const float* __restrict__ bB,
    float* outA_f, float* outB_f, __bf16* outA_bf, __bf16* outB_bf,
    __bf16* outA_bfT, __bf16* outB_bfT) {
  __shared__ __bf16 sA[2 * 64 * 608];  // double-buffered [x(304) | h(304)]
  __shared__ float sC[64 * 304];       // cell state

  int seq = blockIdx.x >> 1;
  int rev = blockIdx.x & 1;
  const __bf16* x = seq ? xB : xA;
  const __bf16* Wt = rev ? wtB : wtF;
  const float* bias = rev ? bB : bF;
  float* out_f = seq ? outB_f : outA_f;
  __bf16* out_bf = seq ? outB_bf : outA_bf;
  __bf16* out_bfT = seq ? outB_bfT : outA_bfT;
  int dir = rev;

  int tid = threadIdx.x;
  int wave = tid >> 5;
  int lane = tid & 31;

  // init h + pad region of both buffers to zero; c to zero
  for (int idx = tid; idx < 64 * 304 * 2; idx += blockDim.x) {
    int buf = idx / (64 * 304);
    int rem = idx % (64 * 304);
    int m = rem / 304, k = rem % 304 + 304;
    sA[buf * (64 * 608) + m * 608 + k] = to_bf(0.f);
  }
  for (int idx = tid; idx < 64 * 304; idx += blockDim.x) sC[idx] = 0.f;

  // prologue: stage x(t0) into buffer 0
  async_copy_x(x, sA, rev ? (Sk - 1) : 0);
  wait_async0();
  __syncthreads();

  for (int ti = 0; ti < Sk; ++ti) {
    int t = rev ? (Sk - 1 - ti) : ti;
    int cur = ti & 1;
    __bf16* bufc = sA + cur * (64 * 608);
    __bf16* bufn = sA + (cur ^ 1) * (64 * 608);

    // overlap: stage x(t+1) into the other buffer while computing step t
    if (ti + 1 < Sk) {
      int tn = rev ? (Sk - 2 - ti) : (ti + 1);
      async_copy_x(x, bufn, tn);
    }

    // 4 M-tiles x 19 N-tiles = 76 tile groups over 16 waves
    for (int g = wave; g < 76; g += 16) {
      int mt = g / 19, nt = g % 19;
      int m0 = mt * 16, n0 = nt * 16;
      int nloc = lane & 15;
      int colH = n0 + nloc;
      v8f acc[4];
#pragma unroll
      for (int q = 0; q < 4; ++q) {
        float bv = (colH < 300) ? bias[q * 300 + colH] : 0.f;
#pragma unroll
        for (int r = 0; r < 8; ++r) acc[q][r] = bv;
      }
      for (int kk = 0; kk < 19; ++kk) {
        v16bf a = load_frag_bf16(bufc, 608, m0, kk * 32);
#pragma unroll
        for (int q = 0; q < 4; ++q) {
          v16bf w = load_frag_bf16(Wt, 608, q * 304 + n0, kk * 32);
          acc[q] = wmma_bf16(a, w, acc[q]);
        }
      }
      // gate nonlinearity + state update (z: i=acc0, j=acc1, f=acc2, o=acc3)
      int mb = (lane >> 4) << 3;
#pragma unroll
      for (int r = 0; r < 8; ++r) {
        int m = m0 + mb + r;
        float zi = acc[0][r], zj = acc[1][r], zf = acc[2][r], zo = acc[3][r];
        float c = sigmoidf_(zf + 1.0f) * sC[m * 304 + colH] +
                  sigmoidf_(zi) * tanhf(zj);
        float h = sigmoidf_(zo) * tanhf(c);
        sC[m * 304 + colH] = c;
        if (colH < 300) {
          bufn[m * 608 + 304 + colH] = to_bf(h);  // h for step t+1
          out_f[((size_t)m * Sk + t) * 600 + dir * 300 + colH] = h;
          if (out_bf)
            out_bf[((size_t)m * Sk + t) * 608 + dir * 300 + colH] = to_bf(h);
          if (out_bfT)
            out_bfT[((size_t)m * 608 + dir * 300 + colH) * Sk + t] = to_bf(h);
        }
      }
    }
    wait_async0();   // x(t+1) resident in LDS
    __syncthreads(); // h(t) writes + step-t reads complete
  }
}

// ---------------------------------------------------------------------------
// sim[b,i,j] = dot(h1[b,i,:600], h2[b,j,:600]); K padded to 608 (zeros).
// One 16x16 tile per wave; 64*256 tiles total.
// ---------------------------------------------------------------------------
__global__ void __launch_bounds__(256) sim_kernel(
    const __bf16* __restrict__ h1, const __bf16* __restrict__ h2,
    float* __restrict__ sim) {
  int wave = threadIdx.x >> 5, lane = threadIdx.x & 31;
  int gid = blockIdx.x * 8 + wave;
  int b = gid >> 8;
  int tt = gid & 255;
  int mt = tt >> 4, nt = tt & 15;
  const __bf16* A = h1 + (size_t)b * Sk * 608;
  const __bf16* Bt = h2 + (size_t)b * Sk * 608;
  v8f acc;
#pragma unroll
  for (int r = 0; r < 8; ++r) acc[r] = 0.f;
  for (int kk = 0; kk < 19; ++kk) {
    v16bf a = load_frag_bf16(A, 608, mt * 16, kk * 32);
    v16bf bb = load_frag_bf16(Bt, 608, nt * 16, kk * 32);
    acc = wmma_bf16(a, bb, acc);
  }
  float* out = sim + (size_t)b * Sk * Sk;
  int n = nt * 16 + (lane & 15);
  int mbase = mt * 16 + ((lane >> 4) << 3);
#pragma unroll
  for (int r = 0; r < 8; ++r) out[(size_t)(mbase + r) * Sk + n] = acc[r];
}

// row softmax of sim -> bf16 probs (for beta)
__global__ void __launch_bounds__(256) softmax_row_kernel(
    const float* __restrict__ sim, __bf16* __restrict__ P) {
  __shared__ float red[256];
  int b = blockIdx.x >> 8, i = blockIdx.x & 255;
  int tid = threadIdx.x;
  float v = sim[((size_t)b * Sk + i) * Sk + tid];
  red[tid] = v;
  __syncthreads();
  for (int s = 128; s > 0; s >>= 1) {
    if (tid < s) red[tid] = fmaxf(red[tid], red[tid + s]);
    __syncthreads();
  }
  float mx = red[0];
  __syncthreads();
  float e = __expf(v - mx);
  red[tid] = e;
  __syncthreads();
  for (int s = 128; s > 0; s >>= 1) {
    if (tid < s) red[tid] += red[tid + s];
    __syncthreads();
  }
  float sum = red[0];
  P[((size_t)b * Sk + i) * Sk + tid] = to_bf(e / sum);
}

// column softmax of sim -> bf16 probs stored TRANSPOSED: P[b][j][i] (alpha)
__global__ void __launch_bounds__(256) softmax_col_kernel(
    const float* __restrict__ sim, __bf16* __restrict__ P) {
  __shared__ float red[256];
  int b = blockIdx.x >> 8, j = blockIdx.x & 255;
  int tid = threadIdx.x;
  float v = sim[((size_t)b * Sk + tid) * Sk + j];
  red[tid] = v;
  __syncthreads();
  for (int s = 128; s > 0; s >>= 1) {
    if (tid < s) red[tid] = fmaxf(red[tid], red[tid + s]);
    __syncthreads();
  }
  float mx = red[0];
  __syncthreads();
  float e = __expf(v - mx);
  red[tid] = e;
  __syncthreads();
  for (int s = 128; s > 0; s >>= 1) {
    if (tid < s) red[tid] += red[tid + s];
    __syncthreads();
  }
  float sum = red[0];
  P[((size_t)b * Sk + j) * Sk + tid] = to_bf(e / sum);
}

// out[b,m,n<600] = sum_k P[b,m,k] * HT[b,n,k]   (K = 256, N tiles = 38)
__global__ void __launch_bounds__(256) attn_gemm_kernel(
    const __bf16* __restrict__ P, const __bf16* __restrict__ HT,
    float* __restrict__ out) {
  int wave = threadIdx.x >> 5, lane = threadIdx.x & 31;
  int gid = blockIdx.x * 8 + wave;          // 64 * 608 tiles
  int b = gid / 608;
  int tt = gid % 608;
  int mt = tt / 38, nt = tt % 38;
  const __bf16* A = P + (size_t)b * Sk * Sk;
  const __bf16* Bt = HT + (size_t)b * 608 * Sk;
  v8f acc;
#pragma unroll
  for (int r = 0; r < 8; ++r) acc[r] = 0.f;
  for (int kk = 0; kk < 8; ++kk) {
    v16bf a = load_frag_bf16(A, Sk, mt * 16, kk * 32);
    v16bf bb = load_frag_bf16(Bt, Sk, nt * 16, kk * 32);
    acc = wmma_bf16(a, bb, acc);
  }
  int n = nt * 16 + (lane & 15);
  int mbase = mt * 16 + ((lane >> 4) << 3);
  if (n < 600) {
#pragma unroll
    for (int r = 0; r < 8; ++r)
      out[((size_t)b * Sk + mbase + r) * 600 + n] = acc[r];
  }
}

// d = relu([h, beta, h*beta, h-beta] @ dec_w + b). A built on the fly (bf16),
// K = 2400 (75 iters), M = B*S = 16384, N padded 304. Wt: (304,2400) bf16.
// Output written directly as bf16 (B,S,304) — the decoder LSTM staging format
// (pad columns are exactly zero because weight/bias pads are zero + relu).
__global__ void __launch_bounds__(256) dec_gemm_kernel(
    const float* __restrict__ hcat, const float* __restrict__ beta,
    const __bf16* __restrict__ Wt, const float* __restrict__ bias,
    __bf16* __restrict__ out) {
  int wave = threadIdx.x >> 5, lane = threadIdx.x & 31;
  int gid = blockIdx.x * 8 + wave;           // 1024 * 19 tiles
  int mt = gid / 19, nt = gid % 19;
  int m0 = mt * 16, n0 = nt * 16;
  int n = n0 + (lane & 15);
  float bv = (n < 300) ? bias[n] : 0.f;
  v8f acc;
#pragma unroll
  for (int r = 0; r < 8; ++r) acc[r] = bv;
  int m = m0 + (lane & 15);
  int koff = frag_koff(lane);
  for (int kk = 0; kk < 75; ++kk) {
    v16bf a;
#pragma unroll
    for (int e = 0; e < 16; ++e) {
      int k = kk * 32 + koff + (e & 7) + ((e >> 3) << 4);
      int seg = (k >= 1800) ? 3 : (k >= 1200) ? 2 : (k >= 600) ? 1 : 0;
      int kkk = k - seg * 600;
      float h = hcat[(size_t)m * 600 + kkk];
      float bt = beta[(size_t)m * 600 + kkk];
      float val = (seg == 0) ? h : (seg == 1) ? bt : (seg == 2) ? h * bt
                                                                : h - bt;
      a[e] = to_bf(val);
    }
    v16bf w = load_frag_bf16(Wt, 2400, n0, kk * 32);
    acc = wmma_bf16(a, w, acc);
  }
  int mbase = m0 + ((lane >> 4) << 3);
#pragma unroll
  for (int r = 0; r < 8; ++r)
    out[(size_t)(mbase + r) * 304 + n] = to_bf(fmaxf(acc[r], 0.f));
}

// pool: v[b] = [sum(g1), max(g1), sum(g2), max(g2)] over seq axis
__global__ void __launch_bounds__(640) pool_kernel(
    const float* __restrict__ g1, const float* __restrict__ g2,
    float* __restrict__ v) {
  int b = blockIdx.x;
  int n = threadIdx.x;
  if (n >= 600) return;
  float s1 = 0.f, m1 = -3.4e38f, s2 = 0.f, m2 = -3.4e38f;
  for (int t = 0; t < Sk; ++t) {
    float a = g1[((size_t)b * Sk + t) * 600 + n];
    s1 += a;
    m1 = fmaxf(m1, a);
    float c = g2[((size_t)b * Sk + t) * 600 + n];
    s2 += c;
    m2 = fmaxf(m2, c);
  }
  float* vb = v + (size_t)b * 2400;
  vb[n] = s1;
  vb[600 + n] = m1;
  vb[1200 + n] = s2;
  vb[1800 + n] = m2;
}

// y = tanh(v @ agg1 + b1); out = y @ agg2 + b2
__global__ void __launch_bounds__(320) agg_kernel(
    const float* __restrict__ v, const float* __restrict__ w1,
    const float* __restrict__ b1, const float* __restrict__ w2,
    const float* __restrict__ b2, float* __restrict__ out) {
  __shared__ float y[300];
  int b = blockIdx.x;
  int j = threadIdx.x;
  if (j < 300) {
    float s = b1[j];
    const float* vb = v + (size_t)b * 2400;
    for (int k = 0; k < 2400; ++k) s += vb[k] * w1[(size_t)k * 300 + j];
    y[j] = tanhf(s);
  }
  __syncthreads();
  if (j < 3) {
    float s = b2[j];
    for (int q = 0; q < 300; ++q) s += y[q] * w2[q * 3 + j];
    out[b * 3 + j] = s;
  }
}

// ---------------------------------------------------------------------------
extern "C" void kernel_launch(void* const* d_in, const int* in_sizes, int n_in,
                              void* d_out, int out_size, void* d_ws,
                              size_t ws_size, hipStream_t stream) {
  (void)in_sizes; (void)n_in; (void)out_size; (void)ws_size;
  const int* x1 = (const int*)d_in[0];
  const int* x2 = (const int*)d_in[1];
  const int* temp1 = (const int*)d_in[2];
  const int* temp2 = (const int*)d_in[3];
  const int* tag1 = (const int*)d_in[4];
  const int* tag2 = (const int*)d_in[5];
  const int* len1 = (const int*)d_in[6];
  const int* len2 = (const int*)d_in[7];
  const float* embed = (const float*)d_in[8];
  const float* tag_c_w = (const float*)d_in[9];
  const float* tag_p_w = (const float*)d_in[10];
  const float* enc_fw_k = (const float*)d_in[11];
  const float* enc_fw_b = (const float*)d_in[12];
  const float* enc_bw_k = (const float*)d_in[13];
  const float* enc_bw_b = (const float*)d_in[14];
  const float* dec_w = (const float*)d_in[15];
  const float* dec_b = (const float*)d_in[16];
  const float* dec_fw_k = (const float*)d_in[17];
  const float* dec_fw_b = (const float*)d_in[18];
  const float* dec_bw_k = (const float*)d_in[19];
  const float* dec_bw_b = (const float*)d_in[20];
  const float* agg1_w = (const float*)d_in[21];
  const float* agg1_b = (const float*)d_in[22];
  const float* agg2_w = (const float*)d_in[23];
  const float* agg2_b = (const float*)d_in[24];
  float* outp = (float*)d_out;

  // ---- workspace bump allocator (256B aligned) ----
  size_t off = 0;
  auto alloc = [&](size_t bytes) -> void* {
    void* p = (char*)d_ws + off;
    off += (bytes + 255) & ~(size_t)255;
    return p;
  };
  const size_t SZ_TREE = (size_t)Bk * Sk * Dk * 4;       // 19.7 MB
  const size_t SZ_WLSTM = (size_t)1216 * 608 * 2;
  const size_t SZ_WDEC = (size_t)304 * 2400 * 2;
  const size_t SZ_HF = (size_t)Bk * Sk * 600 * 4;
  const size_t SZ_HBF = (size_t)Bk * Sk * 608 * 2;
  const size_t SZ_HTBF = (size_t)Bk * 608 * Sk * 2;
  const size_t SZ_SIM = (size_t)Bk * Sk * Sk * 4;
  const size_t SZ_P = (size_t)Bk * Sk * Sk * 2;
  const size_t SZ_XBF = (size_t)Bk * Sk * 304 * 2;

  float* tree1 = (float*)alloc(SZ_TREE);
  float* tree2 = (float*)alloc(SZ_TREE);
  __bf16* wt_enc_fw = (__bf16*)alloc(SZ_WLSTM);
  __bf16* wt_enc_bw = (__bf16*)alloc(SZ_WLSTM);
  __bf16* wt_dec_fw = (__bf16*)alloc(SZ_WLSTM);
  __bf16* wt_dec_bw = (__bf16*)alloc(SZ_WLSTM);
  __bf16* wt_dec = (__bf16*)alloc(SZ_WDEC);
  __bf16* xbf1 = (__bf16*)alloc(SZ_XBF);
  __bf16* xbf2 = (__bf16*)alloc(SZ_XBF);
  float* h1f = (float*)alloc(SZ_HF);
  float* h2f = (float*)alloc(SZ_HF);
  __bf16* h1bf = (__bf16*)alloc(SZ_HBF);   // contiguous group for one zero-fill
  __bf16* h2bf = (__bf16*)alloc(SZ_HBF);
  __bf16* h1tbf = (__bf16*)alloc(SZ_HTBF);
  __bf16* h2tbf = (__bf16*)alloc(SZ_HTBF);
  float* sim = (float*)alloc(SZ_SIM);
  __bf16* Pb = (__bf16*)alloc(SZ_P);
  __bf16* Pa = (__bf16*)alloc(SZ_P);
  float* beta = (float*)alloc(SZ_HF);
  float* alpha = (float*)alloc(SZ_HF);
  __bf16* d1bf = (__bf16*)alloc(SZ_XBF);
  __bf16* d2bf = (__bf16*)alloc(SZ_XBF);
  float* g1f = (float*)alloc(SZ_HF);
  float* g2f = (float*)alloc(SZ_HF);
  float* vpool = (float*)alloc((size_t)Bk * 2400 * 4);

  // ---- pipeline ----
  // zero the bf16 h buffers (K/N padding relies on zeros)
  zero_bf16_kernel<<<2048, 256, 0, stream>>>(h1bf,
      (size_t)2 * (SZ_HBF / 2) + (size_t)2 * (SZ_HTBF / 2));

  conv_lstm_w_kernel<<<1024, 256, 0, stream>>>(enc_fw_k, wt_enc_fw);
  conv_lstm_w_kernel<<<1024, 256, 0, stream>>>(enc_bw_k, wt_enc_bw);
  conv_lstm_w_kernel<<<1024, 256, 0, stream>>>(dec_fw_k, wt_dec_fw);
  conv_lstm_w_kernel<<<1024, 256, 0, stream>>>(dec_bw_k, wt_dec_bw);
  conv_dec_w_kernel<<<1024, 256, 0, stream>>>(dec_w, wt_dec);

  embed_tree_kernel<<<Bk * Sk, 128, 0, stream>>>(x1, embed, tree1);
  embed_tree_kernel<<<Bk * Sk, 128, 0, stream>>>(x2, embed, tree2);
  unfold_kernel<<<Bk, 320, 0, stream>>>(tree1, temp1, tag1, len1, tag_c_w,
                                        tag_p_w);
  unfold_kernel<<<Bk, 320, 0, stream>>>(tree2, temp2, tag2, len2, tag_c_w,
                                        tag_p_w);
  conv_x_kernel<<<1024, 256, 0, stream>>>(tree1, xbf1);
  conv_x_kernel<<<1024, 256, 0, stream>>>(tree2, xbf2);

  // encoder BiLSTM: 4 passes in parallel (seq{1,2} x {fw,bw})
  lstm_kernel<<<4, 512, 0, stream>>>(xbf1, xbf2, wt_enc_fw, enc_fw_b,
                                     wt_enc_bw, enc_bw_b, h1f, h2f, h1bf, h2bf,
                                     h1tbf, h2tbf);

  sim_kernel<<<2048, 256, 0, stream>>>(h1bf, h2bf, sim);
  softmax_row_kernel<<<Bk * Sk, 256, 0, stream>>>(sim, Pb);
  softmax_col_kernel<<<Bk * Sk, 256, 0, stream>>>(sim, Pa);
  attn_gemm_kernel<<<4864, 256, 0, stream>>>(Pb, h2tbf, beta);   // beta
  attn_gemm_kernel<<<4864, 256, 0, stream>>>(Pa, h1tbf, alpha);  // alpha

  dec_gemm_kernel<<<2432, 256, 0, stream>>>(h1f, beta, wt_dec, dec_b, d1bf);
  dec_gemm_kernel<<<2432, 256, 0, stream>>>(h2f, alpha, wt_dec, dec_b, d2bf);

  // decoder BiLSTM (no bf16 copies needed)
  lstm_kernel<<<4, 512, 0, stream>>>(d1bf, d2bf, wt_dec_fw, dec_fw_b,
                                     wt_dec_bw, dec_bw_b, g1f, g2f, nullptr,
                                     nullptr, nullptr, nullptr);

  pool_kernel<<<Bk, 640, 0, stream>>>(g1f, g2f, vpool);
  agg_kernel<<<Bk, 320, 0, stream>>>(vpool, agg1_w, agg1_b, agg2_w, agg2_b,
                                     outp);
}
